// WindowAttention_52845277610718
// MI455X (gfx1250) — matrix operand order repaired and checked by reading, not compiled, not run on hardware
//
#include <hip/hip_runtime.h>
#include <hip/hip_bf16.h>

// ---------------- problem constants ----------------
#define BATCH 32
#define DIM 256
#define HH 64
#define WW 64
#define WIN 8
#define SHIFT 4
#define HEADS 8
#define HEAD_DIM 32
#define NTOK 64                 // WIN*WIN tokens per window
#define NWIN_SIDE 8             // 64/8 windows per side
#define SCALE 0.17677669529663687f   // HEAD_DIM^-0.5

typedef __attribute__((ext_vector_type(16))) _Float16 v16h;
typedef __attribute__((ext_vector_type(8)))  _Float16 v8h;
typedef __attribute__((ext_vector_type(8)))  float    v8f;

// ---------------- LDS layout (bytes) ----------------
// lds_x  : [64][264] f16  (window activations, padded)        33792
// lds_q  : [8][64][40] f16 (per-head q, row-major, padded)    40960
// lds_k  : [8][64][40] f16                                    40960
// lds_vT : [8][32][72] f16 (v transposed: [dim][token])       36864
// lds_p  : [8][64][72] f16 (softmax probs; reused as f32 out) 73728
#define OFF_X   0
#define OFF_Q   33792
#define OFF_K   74752
#define OFF_VT  115712
#define OFF_P   152576
#define LDS_BYTES 226304

#define XPITCH 264
#define QKPITCH 40
#define VPITCH 72
#define PPITCH 72
#define OPITCH 36   // f32 pitch when lds_p reused for output staging

// load 16 f16 from two contiguous-8 chunks (A-fragment style)
__device__ __forceinline__ v16h ld16_2(const _Float16* p0, const _Float16* p1) {
  v8h lo = *(const v8h*)p0;
  v8h hi = *(const v8h*)p1;
  v16h r;
#pragma unroll
  for (int i = 0; i < 8; ++i) { r[i] = lo[i]; r[i + 8] = hi[i]; }
  return r;
}
// load 16 contiguous f16
__device__ __forceinline__ v16h ld16_c(const _Float16* p) { return ld16_2(p, p + 8); }

__device__ __forceinline__ int region_id(int t, int wi, int wj) {
  // token t in window (wi,wj); region boundaries on unshifted grid at 56 and 60
  int ph = wi * WIN + (t >> 3);
  int pw = wj * WIN + (t & 7);
  int rh = (ph < HH - WIN) ? 0 : ((ph < HH - SHIFT) ? 1 : 2);
  int rw = (pw < WW - WIN) ? 0 : ((pw < WW - SHIFT) ? 1 : 2);
  return rh * 3 + rw;
}

// ---------------- pre-pass: qkv_w (256x768 f32) -> Wt (768x256 f16, transposed) ----------------
__global__ void swin_wcast_kernel(const float* __restrict__ qkv_w,
                                  _Float16* __restrict__ Wt) {
  int idx = blockIdx.x * 256 + threadIdx.x;   // 768*256 elements
  int c = idx >> 8;        // output column 0..767
  int k = idx & 255;       // input dim 0..255
  Wt[c * 256 + k] = (_Float16)qkv_w[k * (3 * DIM) + c];
}

// ---------------- fused shifted-window attention ----------------
__global__ __launch_bounds__(256, 1)
void swin_attn_kernel(const float* __restrict__ x,
                      const float* __restrict__ qkv_b,
                      const float* __restrict__ rel_tab,   // [(2W-1)^2][HEADS]
                      const _Float16* __restrict__ Wt,     // [768][256] f16
                      float* __restrict__ out) {
  extern __shared__ char smem[];
  _Float16* lds_x  = (_Float16*)(smem + OFF_X);
  _Float16* lds_q  = (_Float16*)(smem + OFF_Q);
  _Float16* lds_k  = (_Float16*)(smem + OFF_K);
  _Float16* lds_vT = (_Float16*)(smem + OFF_VT);
  _Float16* lds_p  = (_Float16*)(smem + OFF_P);

  const int wid = blockIdx.x;              // 0..2047
  const int b  = wid >> 6;
  const int wi = (wid >> 3) & 7;
  const int wj = wid & 7;

  const int wave = threadIdx.x >> 5;       // 0..7 == head
  const int lane = threadIdx.x & 31;
  const int l16  = lane & 15;
  const bool hi  = lane >= 16;

  // ---- Stage A: gather shifted window -> LDS f16 (streaming read, no reuse) ----
#pragma unroll 4
  for (int idx = threadIdx.x; idx < NTOK * DIM; idx += 256) {
    int t = idx & 63;
    int c = idx >> 6;
    int hs = (wi * WIN + (t >> 3) + SHIFT) & 63;
    int ws = (wj * WIN + (t & 7) + SHIFT) & 63;
    float xv = __builtin_nontemporal_load(&x[((b * DIM + c) * HH + hs) * WW + ws]);
    lds_x[t * XPITCH + c] = (_Float16)xv;
  }
  __syncthreads();

  // ---- Stage B: QKV projection. Wave `wave` == head `wave`.
  // All 24 output tiles (6 column groups x 4 m-tiles) in registers; per k-step,
  // 4 A fragments (LDS, once) and all 6 B fragments (global, one clause off a
  // single base pointer with compile-time immediate offsets).
  _Float16* qh = lds_q + wave * NTOK * QKPITCH;
  _Float16* kh = lds_k + wave * NTOK * QKPITCH;
  _Float16* vh = lds_vT + wave * HEAD_DIM * VPITCH;

  const _Float16* wbase = Wt + (wave * HEAD_DIM + l16) * 256 + (hi ? 8 : 0);
  const _Float16* abase = lds_x + l16 * XPITCH + (hi ? 8 : 0);

  v8f acc[6][4];
#pragma unroll
  for (int ng = 0; ng < 6; ++ng)
#pragma unroll
    for (int mi = 0; mi < 4; ++mi) acc[ng][mi] = v8f{};

#pragma unroll
  for (int k0 = 0; k0 < DIM; k0 += 32) {
    v16h af[4];
#pragma unroll
    for (int mi = 0; mi < 4; ++mi) {
      const _Float16* ap = abase + mi * 16 * XPITCH + k0;
      af[mi] = ld16_2(ap, ap + 16);
    }
    v16h bf[6];
#pragma unroll
    for (int ng = 0; ng < 6; ++ng) {
      // offset is a compile-time constant per unrolled (ng,k0)
      bf[ng] = ld16_2(wbase + ((ng >> 1) * DIM + (ng & 1) * 16) * 256 + k0, wbase + ((ng >> 1) * DIM + (ng & 1) * 16) * 256 + k0 + 16);
    }
#pragma unroll
    for (int ng = 0; ng < 6; ++ng) {
#pragma unroll
      for (int mi = 0; mi < 4; ++mi) {
        acc[ng][mi] = __builtin_amdgcn_wmma_f32_16x16x32_f16(false, af[mi], false, bf[ng],
                                                             (short)0, acc[ng][mi], false, false);
      }
    }
  }

  // epilogue: bias (+ q scale), f16 convert, place q/k row-major, v transposed
#pragma unroll
  for (int ng = 0; ng < 6; ++ng) {
    const int which = ng >> 1;
    const int sub   = ng & 1;
    const int col0  = which * DIM + wave * HEAD_DIM + sub * 16;
    const float bias = qkv_b[col0 + l16];
    const float scl = (which == 0) ? SCALE : 1.0f;
#pragma unroll
    for (int mi = 0; mi < 4; ++mi) {
#pragma unroll
      for (int r = 0; r < 8; ++r) {
        int m = mi * 16 + r + (hi ? 8 : 0);
        _Float16 hval = (_Float16)((acc[ng][mi][r] + bias) * scl);
        int d = sub * 16 + l16;                 // 0..31 within head
        if (which == 0)       qh[m * QKPITCH + d] = hval;
        else if (which == 1)  kh[m * QKPITCH + d] = hval;
        else                  vh[d * VPITCH + m] = hval;   // transposed
      }
    }
  }
  // stage C reads only this wave's own q/k/v -> no inter-wave barrier needed

  // ---- Stage C: S = q * k^T  (64x64, K=32) ----
  v16h aq[4];
#pragma unroll
  for (int mi = 0; mi < 4; ++mi) {
    const _Float16* ap = qh + (mi * 16 + l16) * QKPITCH + (hi ? 8 : 0);
    aq[mi] = ld16_2(ap, ap + 16);
  }
  v8f S[4][4];
#pragma unroll
  for (int nj = 0; nj < 4; ++nj) {
    const _Float16* bp = kh + (nj * 16 + l16) * QKPITCH + (hi ? 8 : 0);
    v16h bk = ld16_2(bp, bp + 16);
#pragma unroll
    for (int mi = 0; mi < 4; ++mi) {
      S[mi][nj] = __builtin_amdgcn_wmma_f32_16x16x32_f16(false, aq[mi], false, bk,
                                                         (short)0, v8f{}, false, false);
    }
  }

  // ---- relative-position bias + shift mask ----
#pragma unroll
  for (int nj = 0; nj < 4; ++nj) {
    const int n = nj * 16 + l16;
    const int idn = region_id(n, wi, wj);
#pragma unroll
    for (int mi = 0; mi < 4; ++mi) {
#pragma unroll
      for (int r = 0; r < 8; ++r) {
        int m = mi * 16 + r + (hi ? 8 : 0);
        int dr = (m >> 3) - (n >> 3) + (WIN - 1);
        int dc = (m & 7) - (n & 7) + (WIN - 1);
        float bv = rel_tab[(dr * (2 * WIN - 1) + dc) * HEADS + wave];
        float mk = (region_id(m, wi, wj) != idn) ? -100.0f : 0.0f;
        S[mi][nj][r] += bv + mk;
      }
    }
  }

  // ---- softmax over keys (rows live across 16-lane halves) ----
#pragma unroll
  for (int mi = 0; mi < 4; ++mi) {
#pragma unroll
    for (int r = 0; r < 8; ++r) {
      float mx = S[mi][0][r];
#pragma unroll
      for (int nj = 1; nj < 4; ++nj) mx = fmaxf(mx, S[mi][nj][r]);
#pragma unroll
      for (int off = 1; off < 16; off <<= 1) mx = fmaxf(mx, __shfl_xor(mx, off, 32));
      float sum = 0.0f;
#pragma unroll
      for (int nj = 0; nj < 4; ++nj) {
        float e = __expf(S[mi][nj][r] - mx);
        S[mi][nj][r] = e;
        sum += e;
      }
#pragma unroll
      for (int off = 1; off < 16; off <<= 1) sum += __shfl_xor(sum, off, 32);
      float is = 1.0f / sum;
#pragma unroll
      for (int nj = 0; nj < 4; ++nj) S[mi][nj][r] *= is;
    }
  }

  // ---- P (f16) -> wave-private LDS for A-fragment reload ----
  _Float16* ph = lds_p + wave * NTOK * PPITCH;
#pragma unroll
  for (int nj = 0; nj < 4; ++nj) {
    int n = nj * 16 + l16;
#pragma unroll
    for (int mi = 0; mi < 4; ++mi) {
#pragma unroll
      for (int r = 0; r < 8; ++r) {
        int m = mi * 16 + r + (hi ? 8 : 0);
        ph[m * PPITCH + n] = (_Float16)S[mi][nj][r];
      }
    }
  }

  // ---- O = P * V  (64x32, K=64): preload all fragments, then 16 WMMAs ----
  {
    const _Float16* pb = ph + l16 * PPITCH + (hi ? 8 : 0);
    const _Float16* vb = vh + l16 * VPITCH + (hi ? 8 : 0);
    v16h apf[4][2], bvf[2][2];
#pragma unroll
    for (int k = 0; k < 2; ++k) {
#pragma unroll
      for (int mi = 0; mi < 4; ++mi) {
        const _Float16* ap = pb + mi * 16 * PPITCH + k * 32;
        apf[mi][k] = ld16_2(ap, ap + 16);
      }
#pragma unroll
      for (int nj = 0; nj < 2; ++nj)
        bvf[nj][k] = ld16_2(vb + nj * 16 * VPITCH + k * 32, vb + nj * 16 * VPITCH + k * 32 + 16);
    }
    v8f O[4][2] = {{v8f{}, v8f{}}, {v8f{}, v8f{}}, {v8f{}, v8f{}}, {v8f{}, v8f{}}};
#pragma unroll
    for (int k = 0; k < 2; ++k)
#pragma unroll
      for (int nj = 0; nj < 2; ++nj)
#pragma unroll
        for (int mi = 0; mi < 4; ++mi)
          O[mi][nj] = __builtin_amdgcn_wmma_f32_16x16x32_f16(false, apf[mi][k], false, bvf[nj][k],
                                                             (short)0, O[mi][nj], false, false);

    // ---- stage O through LDS (reuse P region as f32 [64][36]) ----
    float* oh = (float*)(lds_p) + wave * NTOK * OPITCH;
#pragma unroll
    for (int nj = 0; nj < 2; ++nj) {
      int d = nj * 16 + l16;
#pragma unroll
      for (int mi = 0; mi < 4; ++mi) {
#pragma unroll
        for (int r = 0; r < 8; ++r) {
          int m = mi * 16 + r + (hi ? 8 : 0);
          oh[m * OPITCH + d] = O[mi][nj][r];
        }
      }
    }
  }
  __syncthreads();

  // ---- coalesced scatter to (B,C,H,W); +SHIFT roll folded in (streaming write) ----
#pragma unroll 4
  for (int idx = threadIdx.x; idx < NTOK * DIM; idx += 256) {
    int t = idx & 63;
    int c = idx >> 6;
    int hd = c >> 5, d = c & 31;
    float val = ((float*)lds_p)[hd * NTOK * OPITCH + t * OPITCH + d];
    int hdst = (wi * WIN + (t >> 3) + SHIFT) & 63;
    int wdst = (wj * WIN + (t & 7) + SHIFT) & 63;
    __builtin_nontemporal_store(val, &out[((b * DIM + c) * HH + hdst) * WW + wdst]);
  }
}

extern "C" void kernel_launch(void* const* d_in, const int* in_sizes, int n_in,
                              void* d_out, int out_size, void* d_ws, size_t ws_size,
                              hipStream_t stream) {
  const float* x        = (const float*)d_in[0];
  const float* qkv_w    = (const float*)d_in[1];
  const float* qkv_b    = (const float*)d_in[2];
  const float* rel_tab  = (const float*)d_in[3];
  float* out            = (float*)d_out;
  _Float16* Wt          = (_Float16*)d_ws;   // 768*256 f16 = 393216 B

  // pre-pass: transpose + f16-cast weights
  swin_wcast_kernel<<<(3 * DIM * DIM) / 256, 256, 0, stream>>>(qkv_w, Wt);

  // fused attention: one workgroup (8 waves) per window
  const int nblocks = BATCH * NWIN_SIDE * NWIN_SIDE;   // 2048
  swin_attn_kernel<<<nblocks, 256, LDS_BYTES, stream>>>(x, qkv_b, rel_tab, Wt, out);
}
